// VanillaRNN_60567628808784
// MI455X (gfx1250) — compile-verified, matrix operand order checked by reading, not run on hardware
//
#include <hip/hip_runtime.h>
#include <hip/hip_bf16.h>
#include <math.h>

// ---------------------------------------------------------------------------
// Stacked Elman RNN LM for MI455X (gfx1250, wave32, WMMA, async-to-LDS).
//   one-shot: wte/W_ih/W_hh fp32 -> bf16 (halves streamed bytes for GEMMs)
//   x_bf = wte_bf[idx]                                (embed gather)
//   per layer: pre = x_bf @ W_ih_bf^T + b_ih + b_hh   (WMMA bf16 GEMM, fp32 acc,
//                double-buffered GLOBAL_LOAD_ASYNC_TO_LDS_B128 staging)
//              h_t = tanh(pre_t + h_{t-1} @ W_hh^T)   (persistent scan kernel,
//                W_hh slice LDS-resident bf16 via async copy, M padded 4->16)
//   xn_bf = LayerNorm(x_f32) * gamma                  (fp32 stats, bf16 out)
//   logits = xn_bf @ wte_bf^T                         (WMMA GEMM, fp32 out)
// ---------------------------------------------------------------------------

#define RNN_V 32000
#define RNN_H 1024
#define RNN_L 8
#define RNN_B 4
#define RNN_T 1024
#define LN_EPS 1e-5f

typedef __attribute__((ext_vector_type(16))) __bf16 v16bf;
typedef __attribute__((ext_vector_type(8)))  __bf16 v8bf;
typedef __attribute__((ext_vector_type(8)))  float  v8f;

// Async copy global->LDS (CDNA5, tracked by ASYNCcnt). VDST is the LDS byte
// address; low 32 bits of a generic pointer into LDS are the LDS address
// (ISA 10.2 aperture mapping).
#define ASYNC_B128(ldsoff, gptr)                                         \
  asm volatile("global_load_async_to_lds_b128 %0, %1, off"               \
               :: "v"(ldsoff), "v"(gptr) : "memory")
#define WAIT_ASYNC() asm volatile("s_wait_asynccnt 0x0" ::: "memory")

static __device__ __forceinline__ unsigned lds_off(const void* p) {
  return (unsigned)(unsigned long long)p;
}

// ---------------------------------------------------------------------------
// One-shot fp32 -> bf16 conversion (n % 8 == 0).
// ---------------------------------------------------------------------------
__global__ __launch_bounds__(256) void k_cvt(const float* __restrict__ s,
                                             __bf16* __restrict__ d, long n) {
  const long i = ((long)blockIdx.x * blockDim.x + threadIdx.x) * 8;
  if (i >= n) return;
  const float4 a = *(const float4*)&s[i];
  const float4 b = *(const float4*)&s[i + 4];
  v8bf o;
  o[0] = (__bf16)a.x; o[1] = (__bf16)a.y; o[2] = (__bf16)a.z; o[3] = (__bf16)a.w;
  o[4] = (__bf16)b.x; o[5] = (__bf16)b.y; o[6] = (__bf16)b.z; o[7] = (__bf16)b.w;
  *(v8bf*)&d[i] = o;
}

// ---------------------------------------------------------------------------
// Embedding gather (bf16 rows, 2 KB each): x_bf[row,:] = wte_bf[idx[row],:]
// ---------------------------------------------------------------------------
__global__ __launch_bounds__(128) void k_embed(const int* __restrict__ idx,
                                               const __bf16* __restrict__ wte_bf,
                                               __bf16* __restrict__ xbf) {
  const int row = blockIdx.x;  // b*T + t
  const int token = idx[row];
  const uint4* src = (const uint4*)(wte_bf + (size_t)token * RNN_H);
  uint4* dst = (uint4*)(xbf + (size_t)row * RNN_H);
  for (int i = threadIdx.x; i < RNN_H / 8; i += blockDim.x) dst[i] = src[i];
}

__global__ void k_zero_bar(int* __restrict__ bar) {
  if (threadIdx.x == 0) *bar = 0;
}

// ---------------------------------------------------------------------------
// C(M,N) = A(M,K) @ Bmat(N,K)^T [+ bias0[n] + bias1[n]], A/B bf16, C fp32.
// Block: 256 threads = 8 waves, tile 128x128, K-step 32, double-buffered LDS
// staged with GLOBAL_LOAD_ASYNC_TO_LDS_B128 (copy next k-tile while WMMAs run
// on the current one; one s_wait_asynccnt + barrier per iteration).
// Requires M%128==0, N%128==0, K%32==0 (M=4096, N in {1024,32000}, K=1024).
// ---------------------------------------------------------------------------
#define GBM 128
#define GBN 128
#define GBK 32
#define GLD 40   // padded LDS row stride (bf16): 80B rows -> conflict-free frags

__global__ __launch_bounds__(256) void k_gemm_bf16(
    const __bf16* __restrict__ A, const __bf16* __restrict__ Bmat,
    const float* __restrict__ bias0, const float* __restrict__ bias1,
    float* __restrict__ C, int M, int N, int K) {
  __shared__ __align__(16) __bf16 As[2][GBM * GLD];
  __shared__ __align__(16) __bf16 Bs[2][GBN * GLD];

  const int tid = threadIdx.x;
  const int m0 = blockIdx.y * GBM;
  const int n0 = blockIdx.x * GBN;

  const int wave = tid >> 5, lane = tid & 31;
  const int wm = wave >> 1;          // 0..3 -> 32 rows each (2 m-tiles)
  const int wn = wave & 1;           // 0..1 -> 64 cols each (4 n-tiles)
  const int lrow = lane & 15;
  const int kb = (lane >> 4) << 3;   // 0 or 8

  const unsigned aoff[2] = {lds_off(&As[0][0]), lds_off(&As[1][0])};
  const unsigned boff[2] = {lds_off(&Bs[0][0]), lds_off(&Bs[1][0])};

  // staging map: per matrix 128 rows x 4 b128-chunks = 512 chunks; 2/thread
  const int srow = tid >> 1;         // 0..127
  const int sc = (tid & 1) * 2;      // chunk base (16B chunks)

  auto stage = [&](int buf, int k0) {
#pragma unroll
    for (int c = sc; c < sc + 2; ++c) {
      ASYNC_B128(aoff[buf] + srow * (GLD * 2) + c * 16,
                 A + (size_t)(m0 + srow) * K + k0 + c * 8);
      ASYNC_B128(boff[buf] + srow * (GLD * 2) + c * 16,
                 Bmat + (size_t)(n0 + srow) * K + k0 + c * 8);
    }
  };

  v8f zero = {};
  v8f acc[2][4];
#pragma unroll
  for (int i = 0; i < 2; ++i)
#pragma unroll
    for (int j = 0; j < 4; ++j) acc[i][j] = zero;

  stage(0, 0);
  WAIT_ASYNC();
  __syncthreads();

  int buf = 0;
  for (int k0 = 0; k0 < K; k0 += GBK) {
    if (k0 + GBK < K) stage(buf ^ 1, k0 + GBK);  // async copy next tile

    const __bf16* as = As[buf];
    const __bf16* bs = Bs[buf];
    v16bf afrag[2], bfrag[4];
#pragma unroll
    for (int mt = 0; mt < 2; ++mt) {
      const int r = (wm * 32 + mt * 16 + lrow) * GLD;
      v8bf lo = *(const v8bf*)&as[r + kb];
      v8bf hi = *(const v8bf*)&as[r + 16 + kb];
#pragma unroll
      for (int j = 0; j < 8; ++j) { afrag[mt][j] = lo[j]; afrag[mt][j + 8] = hi[j]; }
    }
#pragma unroll
    for (int nt = 0; nt < 4; ++nt) {
      const int r = (wn * 64 + nt * 16 + lrow) * GLD;
      v8bf lo = *(const v8bf*)&bs[r + kb];
      v8bf hi = *(const v8bf*)&bs[r + 16 + kb];
#pragma unroll
      for (int j = 0; j < 8; ++j) { bfrag[nt][j] = lo[j]; bfrag[nt][j + 8] = hi[j]; }
    }
#pragma unroll
    for (int mt = 0; mt < 2; ++mt)
#pragma unroll
      for (int nt = 0; nt < 4; ++nt)
        acc[mt][nt] = __builtin_amdgcn_wmma_f32_16x16x32_bf16(
            false, afrag[mt], false, bfrag[nt], (short)0, acc[mt][nt], false, false);

    WAIT_ASYNC();      // next tile resident
    __syncthreads();   // everyone done reading current tile
    buf ^= 1;
  }

  // epilogue: D layout -> VGPR r: lane<16 (M=r, N=lane), lane>=16 (M=r+8, N=lane-16)
  const int mhi = (lane >> 4) << 3;
#pragma unroll
  for (int mt = 0; mt < 2; ++mt)
#pragma unroll
    for (int nt = 0; nt < 4; ++nt) {
      const int gn = n0 + wn * 64 + nt * 16 + lrow;
      float badd = (bias0 != nullptr) ? (bias0[gn] + bias1[gn]) : 0.0f;
#pragma unroll
      for (int r = 0; r < 8; ++r) {
        const int gm = m0 + wm * 32 + mt * 16 + mhi + r;
        C[(size_t)gm * N + gn] = acc[mt][nt][r] + badd;
      }
    }
}

// ---------------------------------------------------------------------------
// Recurrent scan for one layer. Grid of 16 persistent WGs x 128 threads.
// Each WG owns 64 output columns; its W_hh^T slice (64 x 1024 bf16, padded to
// 64*1032*2B = 129 KB -- CDNA5 320KB LDS) is staged with async b128 copies.
// Each wave owns one 16-column tile. Per step: C := pre_t (WMMA accumulator),
// 32 WMMA k-steps against h_{t-1} (global bf16 ping-pong), tanh, write
// fp32 + bf16 layer output and next h. Grid barrier between steps.
// ---------------------------------------------------------------------------
#define SCAN_WGS 16
#define SCAN_KP 1032  // padded column stride (bf16); 2064B, 16B aligned

__global__ __launch_bounds__(128) void k_scan(
    const __bf16* __restrict__ Whh_bf,  // (H, H) row n = output col, contig k
    const float* __restrict__ pre,      // (B, T, H)
    float* __restrict__ xout,           // (B, T, H) layer output fp32 (for LN)
    __bf16* __restrict__ xbf,           // (B, T, H) layer output bf16 (next GEMM)
    __bf16* __restrict__ hbf,           // 2 x (B, H) bf16 ping-pong h state
    int* __restrict__ bar) {
  extern __shared__ __align__(16) __bf16 Ws[];  // [64][SCAN_KP]

  const int tid = threadIdx.x;
  const int colbase = blockIdx.x * 64;
  const unsigned wsoff = lds_off(&Ws[0]);

  // Async-stage W_hh slice: 64 rows x 128 b128-chunks (2KB data + 16B pad/row).
  for (int i = tid; i < 64 * 128; i += 128) {
    const int r = i >> 7;
    const int c = i & 127;
    ASYNC_B128(wsoff + r * (SCAN_KP * 2) + c * 16,
               Whh_bf + (size_t)(colbase + r) * RNN_H + c * 8);
  }
  WAIT_ASYNC();
  __syncthreads();

  const int wave = tid >> 5, lane = tid & 31;
  const int lrow = lane & 15;
  const int kb = (lane >> 4) << 3;             // 0 or 8
  const int colloc = wave * 16 + lrow;         // column within LDS slice
  const int mycol = colbase + colloc;          // global output column

  for (int t = 0; t < RNN_T; ++t) {
    // C accumulator seeded with pre_t (rows M=0..3 = batch; rest zero-pad)
    v8f c = {};
    if (lane < 16) {
#pragma unroll
      for (int r = 0; r < RNN_B; ++r)
        c[r] = pre[((size_t)r * RNN_T + t) * RNN_H + mycol];
    }

    if (t > 0) {
      const __bf16* hsrc = hbf + ((t + 1) & 1) * (RNN_B * RNN_H);  // h_{t-1}
#pragma unroll 4
      for (int ks = 0; ks < RNN_H / 32; ++ks) {
        const int k0 = ks * 32;
        // A fragment: 16x32 bf16, rows 0..3 = h_{t-1}, rows 4..15 = 0
        v8bf alo = {}, ahi = {};
        if (lrow < RNN_B) {
          alo = *(const v8bf*)&hsrc[lrow * RNN_H + k0 + kb];
          ahi = *(const v8bf*)&hsrc[lrow * RNN_H + k0 + 16 + kb];
        }
        v16bf a, b;
#pragma unroll
        for (int j = 0; j < 8; ++j) { a[j] = alo[j]; a[j + 8] = ahi[j]; }
        // B fragment: W_hh^T tile from LDS, B[k][n] = Whh[n][k]
        const v8bf blo = *(const v8bf*)&Ws[colloc * SCAN_KP + k0 + kb];
        const v8bf bhi = *(const v8bf*)&Ws[colloc * SCAN_KP + k0 + 16 + kb];
#pragma unroll
        for (int j = 0; j < 8; ++j) { b[j] = blo[j]; b[j + 8] = bhi[j]; }
        c = __builtin_amdgcn_wmma_f32_16x16x32_bf16(
            false, a, false, b, (short)0, c, false, false);
      }
    }

    // h_t = tanh(.) for the 4 real batch rows; write outputs + next state
    if (lane < 16) {
      __bf16* hdst = hbf + (t & 1) * (RNN_B * RNN_H);
#pragma unroll
      for (int r = 0; r < RNN_B; ++r) {
        const float v = tanhf(c[r]);
        const size_t o = ((size_t)r * RNN_T + t) * RNN_H + mycol;
        xout[o] = v;
        xbf[o] = (__bf16)v;
        hdst[r * RNN_H + mycol] = (__bf16)v;
      }
    }

    // grid-wide barrier: release writes, arrive, spin with s_sleep backoff
    __threadfence();
    __syncthreads();
    if (tid == 0) {
      atomicAdd(bar, 1);
      const int target = SCAN_WGS * (t + 1);
      while (__hip_atomic_load(bar, __ATOMIC_ACQUIRE, __HIP_MEMORY_SCOPE_AGENT) < target)
        __builtin_amdgcn_s_sleep(2);
    }
    __syncthreads();
    __threadfence();
  }
}

// ---------------------------------------------------------------------------
// LayerNorm (no bias): fp32 stats, bf16 output for the LM-head GEMM.
// One 256-thread block per row, H=1024, wave32 shuffles.
// ---------------------------------------------------------------------------
__global__ __launch_bounds__(256) void k_ln(const float* __restrict__ x,
                                            const float* __restrict__ gamma,
                                            __bf16* __restrict__ xn) {
  const int row = blockIdx.x;
  const int tid = threadIdx.x;
  const float* xr = x + (size_t)row * RNN_H;
  float v[4], s = 0.f, s2 = 0.f;
#pragma unroll
  for (int i = 0; i < 4; ++i) {
    v[i] = xr[tid + i * 256];
    s += v[i];
    s2 += v[i] * v[i];
  }
#pragma unroll
  for (int off = 16; off > 0; off >>= 1) {
    s  += __shfl_xor(s,  off, 32);
    s2 += __shfl_xor(s2, off, 32);
  }
  __shared__ float ps[8], ps2[8];
  if ((tid & 31) == 0) { ps[tid >> 5] = s; ps2[tid >> 5] = s2; }
  __syncthreads();
  s = 0.f; s2 = 0.f;
#pragma unroll
  for (int w = 0; w < 8; ++w) { s += ps[w]; s2 += ps2[w]; }
  const float mu = s * (1.0f / RNN_H);
  const float var = s2 * (1.0f / RNN_H) - mu * mu;
  const float rs = rsqrtf(var + LN_EPS);
  __bf16* xo = xn + (size_t)row * RNN_H;
#pragma unroll
  for (int i = 0; i < 4; ++i)
    xo[tid + i * 256] = (__bf16)((v[i] - mu) * rs * gamma[tid + i * 256]);
}

// ---------------------------------------------------------------------------
// Orchestration. Inputs (setup_inputs order):
//   0: idx (B,T) int32        1: wte (V,H) f32      2: W_ih (L,H,H) f32
//   3: W_hh (L,H,H) f32       4: bias_ih (L,H) f32  5: bias_hh (L,H) f32
//   6: gamma (H) f32
// Output: logits (B,T,V) f32.
// Workspace (~144 MB): bar | hbf | x_f32(16M) | pre(16M) | x_bf(8M) |
//   xn_bf(8M) | wte_bf(64M) | Wih_bf(16M) | Whh_bf(16M)
// ---------------------------------------------------------------------------
extern "C" void kernel_launch(void* const* d_in, const int* in_sizes, int n_in,
                              void* d_out, int out_size, void* d_ws, size_t ws_size,
                              hipStream_t stream) {
  (void)in_sizes; (void)n_in; (void)out_size; (void)ws_size;
  const int*   idx   = (const int*)d_in[0];
  const float* wte   = (const float*)d_in[1];
  const float* W_ih  = (const float*)d_in[2];
  const float* W_hh  = (const float*)d_in[3];
  const float* b_ih  = (const float*)d_in[4];
  const float* b_hh  = (const float*)d_in[5];
  const float* gamma = (const float*)d_in[6];
  float* out = (float*)d_out;

  const size_t BTH = (size_t)RNN_B * RNN_T * RNN_H;   // 4M elems
  const size_t LHH = (size_t)RNN_L * RNN_H * RNN_H;   // 8M elems
  const size_t VH  = (size_t)RNN_V * RNN_H;           // 32.768M elems

  char* ws = (char*)d_ws;
  size_t off = 0;
  int*    bar    = (int*)(ws + off);     off += 256;
  __bf16* hbf    = (__bf16*)(ws + off);  off += 2 * RNN_B * RNN_H * sizeof(__bf16);
  off = (off + 255) & ~(size_t)255;
  float*  x_f32  = (float*)(ws + off);   off += BTH * sizeof(float);
  float*  pre    = (float*)(ws + off);   off += BTH * sizeof(float);
  __bf16* x_bf   = (__bf16*)(ws + off);  off += BTH * sizeof(__bf16);
  __bf16* xn_bf  = (__bf16*)(ws + off);  off += BTH * sizeof(__bf16);
  __bf16* wte_bf = (__bf16*)(ws + off);  off += VH * sizeof(__bf16);
  __bf16* Wih_bf = (__bf16*)(ws + off);  off += LHH * sizeof(__bf16);
  __bf16* Whh_bf = (__bf16*)(ws + off);  off += LHH * sizeof(__bf16);

  const int M = RNN_B * RNN_T;  // 4096

  // One-shot bf16 conversions (wte fp32 no longer touched by the GEMMs).
  k_cvt<<<(int)(VH / 2048),  256, 0, stream>>>(wte,  wte_bf, (long)VH);
  k_cvt<<<(int)(LHH / 2048), 256, 0, stream>>>(W_ih, Wih_bf, (long)LHH);
  k_cvt<<<(int)(LHH / 2048), 256, 0, stream>>>(W_hh, Whh_bf, (long)LHH);

  k_embed<<<M, 128, 0, stream>>>(idx, wte_bf, x_bf);

  const size_t scan_lds = (size_t)64 * SCAN_KP * sizeof(__bf16);  // ~129 KB
  for (int l = 0; l < RNN_L; ++l) {
    // pre = x @ W_ih[l]^T + b_ih[l] + b_hh[l]
    k_gemm_bf16<<<dim3(RNN_H / GBN, M / GBM), 256, 0, stream>>>(
        x_bf, Wih_bf + (size_t)l * RNN_H * RNN_H,
        b_ih + (size_t)l * RNN_H, b_hh + (size_t)l * RNN_H,
        pre, M, RNN_H, RNN_H);
    k_zero_bar<<<1, 64, 0, stream>>>(bar);
    k_scan<<<SCAN_WGS, 128, scan_lds, stream>>>(
        Whh_bf + (size_t)l * RNN_H * RNN_H, pre, x_f32, x_bf, hbf, bar);
  }

  k_ln<<<M, 256, 0, stream>>>(x_f32, gamma, xn_bf);

  // logits = xn @ wte^T  (M=4096, N=32000, K=1024; wte_bf (64MB) lives in L2)
  k_gemm_bf16<<<dim3(RNN_V / GBN, M / GBM), 256, 0, stream>>>(
      xn_bf, wte_bf, nullptr, nullptr, out, M, RNN_V, RNN_H);
}
